// MultiheadDotProductAttention_37434934952242
// MI455X (gfx1250) — matrix operand order repaired and checked by reading, not compiled, not run on hardware
//
#include <hip/hip_runtime.h>
#include <hip/hip_bf16.h>
#include <cstdint>

// ---------------------------------------------------------------------------
// MI455X (gfx1250) fused multi-head attention forward.
// B=4, L=2048, E=1024, H=16, D=64.  Output = (o [B,L,E] fp32, attention
// [B,H,L,L] fp32) concatenated in d_out.  Matmuls: v_wmma_f32_16x16x32_f16.
// Tile staging: GLOBAL_LOAD_ASYNC_TO_LDS_B128 + s_wait_asynccnt, double
// buffered.
// ---------------------------------------------------------------------------

typedef _Float16 h16;
typedef __attribute__((ext_vector_type(16))) _Float16 v16h;
typedef __attribute__((ext_vector_type(4)))  _Float16 v4h;
typedef __attribute__((ext_vector_type(8)))  float    v8f;
typedef __attribute__((ext_vector_type(4)))  float    v4f;
typedef __attribute__((ext_vector_type(4)))  int      v4i;

#define B_   4
#define L_   2048
#define E_   1024
#define H_   16
#define D_   64
#define M_   (B_ * L_)          // 8192

// ---- async global->LDS copy (16 bytes per lane) ----------------------------
#if defined(__HIP_DEVICE_COMPILE__) && \
    __has_builtin(__builtin_amdgcn_global_load_async_to_lds_b128)
#define HAVE_ASYNC 1
#else
#define HAVE_ASYNC 0
#endif

__device__ __forceinline__ void async_cp16(const void* g, void* l) {
#if HAVE_ASYNC
  __builtin_amdgcn_global_load_async_to_lds_b128((v4i*)g, (v4i*)l, 0, 0);
#else
  const unsigned long long* gs = (const unsigned long long*)g;
  unsigned long long* ls = (unsigned long long*)l;
  ls[0] = gs[0]; ls[1] = gs[1];
#endif
}

__device__ __forceinline__ void async_wait0() {
#if HAVE_ASYNC
#if __has_builtin(__builtin_amdgcn_s_wait_asynccnt)
  __builtin_amdgcn_s_wait_asynccnt(0);
#else
  asm volatile("s_wait_asynccnt 0x0" ::: "memory");
#endif
#endif
}

// ---- WMMA fragment helpers (layouts per CDNA5 ISA 7.12.2, wave32) ----------

// A matrix 16x32 f16: lane(0..15): row M=lane, K = 0..7 / 16..23;
// lanes 16..31: row M=lane-16, K = 8..15 / 24..31.
__device__ __forceinline__ v16h lds_a_frag(const h16* __restrict__ p0,
                                           int stride, int lane) {
  const h16* p = p0 + (size_t)(lane & 15) * stride + ((lane >> 4) << 3);
  v16h a;
#pragma unroll
  for (int i = 0; i < 8; ++i) { a[i] = p[i]; a[i + 8] = p[16 + i]; }
  return a;
}

// B matrix 32x16 f16 from LDS tile stored transposed Bt[n][k]:
// lane(0..15): N=lane, K=0..7 / 16..23; lanes 16..31: N=lane-16, K=8..15 / 24..31.
__device__ __forceinline__ v16h lds_b_frag(const h16* __restrict__ p0,
                                           int stride, int lane) {
  const h16* p = p0 + (size_t)(lane & 15) * stride + ((lane >> 4) << 3);
  v16h b;
#pragma unroll
  for (int i = 0; i < 8; ++i) { b[i] = p[i]; b[i + 8] = p[16 + i]; }
  return b;
}

__device__ __forceinline__ v8f wmma_f16(v16h a, v16h b, v8f c) {
  return __builtin_amdgcn_wmma_f32_16x16x32_f16(false, a, false, b,
                                                (short)0, c, false, false);
}

__device__ __forceinline__ v8f vzero8() {
  v8f z = {0.f, 0.f, 0.f, 0.f, 0.f, 0.f, 0.f, 0.f};
  return z;
}

// ---------------------------------------------------------------------------
// Weight transpose + convert: Wt[n][k] = (f16) W[k][n]   (1024x1024)
// ---------------------------------------------------------------------------
__global__ __launch_bounds__(256)
void transpose_w_kernel(const float* __restrict__ W, h16* __restrict__ Wt) {
  __shared__ float s[32 * 33];
  const int tid = threadIdx.x;
  const int n0 = blockIdx.x * 32, k0 = blockIdx.y * 32;
  for (int i = tid; i < 32 * 32; i += 256) {
    int r = i >> 5, c = i & 31;
    s[r * 33 + c] = W[(size_t)(k0 + r) * E_ + n0 + c];
  }
  __syncthreads();
  for (int i = tid; i < 32 * 32; i += 256) {
    int r = i >> 5, c = i & 31;
    Wt[(size_t)(n0 + r) * E_ + k0 + c] = (h16)s[c * 33 + r];
  }
}

// ---------------------------------------------------------------------------
// GEMM:  Y = (A @ W + bias) * scale   with Wt = W^T in f16
//   outMode 0: A fp32 [M,K], out f16 head layout [B,H,L,D]      (Q, K)
//   outMode 2: A fp32 [M,K], out f16 transposed  [B,H,D,L]      (V)
//   outMode 1: A f16  [M,K], out fp32 row-major  [M,N]          (o)
// Block tile 128x64, BK=64 (8 WMMA / stage), double-buffered async staging.
// ---------------------------------------------------------------------------
#define BKG 64
#define AS_STRIDE 72   // 64 + pad (144B rows: 16 distinct banks, 16B aligned)
#define BT_STRIDE 72

__global__ __launch_bounds__(256)
void gemm_kernel(const float* __restrict__ A32, const h16* __restrict__ A16,
                 const h16* __restrict__ Wt, const float* __restrict__ bias,
                 h16* __restrict__ out16, float* __restrict__ out32,
                 int M, int N, int K, float scale, int outMode) {
  __shared__ h16 As[2][128 * AS_STRIDE];
  __shared__ h16 Bt[2][64 * BT_STRIDE];

  const int tid  = threadIdx.x;
  const int lane = tid & 31;
  const int w    = tid >> 5;
  const int m0   = blockIdx.y * 128;
  const int n0   = blockIdx.x * 64;
  const int wm   = (w >> 1) * 32;
  const int wn   = (w & 1) * 32;
  const bool aF16 = (outMode == 1);

  auto stage = [&](int buf, int k0) {
    // B tile: 64 n-rows x 64 k = 512 x 16B chunks
#pragma unroll
    for (int j = 0; j < 2; ++j) {
      int c = j * 256 + tid;
      int n = c >> 3, c8 = c & 7;
      async_cp16(Wt + (size_t)(n0 + n) * K + k0 + c8 * 8,
                 &Bt[buf][n * BT_STRIDE + c8 * 8]);
    }
    if (aF16) {  // A tile: 128 rows x 64 k = 1024 x 16B chunks
#pragma unroll
      for (int j = 0; j < 4; ++j) {
        int c = j * 256 + tid;
        int r = c >> 3, c8 = c & 7;
        async_cp16(A16 + (size_t)(m0 + r) * K + k0 + c8 * 8,
                   &As[buf][r * AS_STRIDE + c8 * 8]);
      }
    } else {     // fp32 -> f16: float4 load, packed cvt, 8B LDS store
#pragma unroll
      for (int j = 0; j < 8; ++j) {
        int c = j * 256 + tid;
        int r = c >> 4, c4 = (c & 15) * 4;
        v4f f = *(const v4f*)(A32 + (size_t)(m0 + r) * K + k0 + c4);
        *(v4h*)&As[buf][r * AS_STRIDE + c4] = __builtin_convertvector(f, v4h);
      }
    }
  };

  v8f acc[2][2];
  acc[0][0] = vzero8(); acc[0][1] = vzero8();
  acc[1][0] = vzero8(); acc[1][1] = vzero8();

  stage(0, 0);
  int buf = 0;
  for (int k0 = 0; k0 < K; k0 += BKG) {
    async_wait0();
    __syncthreads();
    if (k0 + BKG < K) stage(buf ^ 1, k0 + BKG);

#pragma unroll
    for (int kk = 0; kk < BKG; kk += 32) {
      v16h a0 = lds_a_frag(&As[buf][(wm + 0)  * AS_STRIDE + kk], AS_STRIDE, lane);
      v16h a1 = lds_a_frag(&As[buf][(wm + 16) * AS_STRIDE + kk], AS_STRIDE, lane);
      v16h b0 = lds_b_frag(&Bt[buf][(wn + 0)  * BT_STRIDE + kk], BT_STRIDE, lane);
      v16h b1 = lds_b_frag(&Bt[buf][(wn + 16) * BT_STRIDE + kk], BT_STRIDE, lane);
      acc[0][0] = wmma_f16(a0, b0, acc[0][0]);
      acc[0][1] = wmma_f16(a0, b1, acc[0][1]);
      acc[1][0] = wmma_f16(a1, b0, acc[1][0]);
      acc[1][1] = wmma_f16(a1, b1, acc[1][1]);
    }
    buf ^= 1;
  }

  // Epilogue. C layout: VGPR r, lanes 0-15 -> M=r, lanes 16-31 -> M=r+8.
  const int colInTile = lane & 15;
  const int rowHalf   = (lane >> 4) << 3;
#pragma unroll
  for (int i = 0; i < 2; ++i) {
#pragma unroll
    for (int j = 0; j < 2; ++j) {
      const int gn = n0 + wn + j * 16 + colInTile;
      const float bv = bias[gn];
#pragma unroll
      for (int r = 0; r < 8; ++r) {
        const int gm = m0 + wm + i * 16 + r + rowHalf;
        const float v = (acc[i][j][r] + bv) * scale;
        const int b = gm >> 11, l = gm & (L_ - 1);
        const int h = gn >> 6,  d = gn & (D_ - 1);
        if (outMode == 0) {
          out16[(((size_t)b * H_ + h) * L_ + l) * D_ + d] = (h16)v;
        } else if (outMode == 2) {
          out16[(((size_t)b * H_ + h) * D_ + d) * L_ + l] = (h16)v;
        } else {
          out32[(size_t)gm * N + gn] = v;
        }
      }
    }
  }
}

// ---------------------------------------------------------------------------
// Fused attention, per (b*h, 16-row q strip):
//   S = Q @ K^T (Q pre-scaled), masked, full 16x2048 fp32 strip in LDS
//   softmax fp32 -> write normalized P to d_out + f16 LDS copy
//   O = P @ V (V pre-transposed [B,H,D,L]), split-K across wave halves
// All K/V/Q tile staging is async + double buffered.
// ---------------------------------------------------------------------------
#define QS_STRIDE 72     // 64 + pad  (144B rows)
#define KS_STRIDE 72     // K tiles [key][d]
#define VT_STRIDE 136    // V tiles [d][key], 128 + pad (272B rows)
#define PS_STRIDE 2056   // 2048 + pad (4112B rows)

#define SS_BYTES   (16 * 2048 * 4)                 // 131072 (also holds Vs)
#define PS_BYTES   (16 * PS_STRIDE * 2)            // 65792
#define QS_BYTES   (16 * QS_STRIDE * 2)            // 2304
#define KV_BYTES   (2 * 128 * KS_STRIDE * 2)       // 36864 (K double buffer)
#define STAT_BYTES (16 * 2 * 4)
#define ATTN_SHMEM (SS_BYTES + PS_BYTES + QS_BYTES + KV_BYTES + STAT_BYTES)

__global__ __launch_bounds__(256)
void attn_kernel(const h16* __restrict__ Q16, const h16* __restrict__ K16,
                 const h16* __restrict__ V16t, const int* __restrict__ mask,
                 float* __restrict__ attOut, h16* __restrict__ VAL16) {
  extern __shared__ char smem[];
  float* Ss     = (float*)smem;                                  // [16][2048]
  h16*   Ps     = (h16*)(smem + SS_BYTES);                       // [16][2056]
  h16*   Qs     = (h16*)(smem + SS_BYTES + PS_BYTES);            // [16][72]
  h16*   KV     = (h16*)(smem + SS_BYTES + PS_BYTES + QS_BYTES); // K x2 buf
  float* rowmax = (float*)(smem + SS_BYTES + PS_BYTES + QS_BYTES + KV_BYTES);
  float* rowinv = rowmax + 16;

  const int tid  = threadIdx.x;
  const int lane = tid & 31;
  const int w    = tid >> 5;
  const int bh   = blockIdx.y;        // b*H + h
  const int b    = bh >> 4;
  const int h    = bh & (H_ - 1);
  const int q0   = blockIdx.x * 16;

  auto stageK = [&](int buf, int kt) {   // 128 keys x 64 d = 1024 chunks
#pragma unroll
    for (int j = 0; j < 4; ++j) {
      int c = j * 256 + tid;
      int key = c >> 3, d8 = c & 7;
      async_cp16(K16 + ((size_t)bh * L_ + kt + key) * D_ + d8 * 8,
                 KV + (size_t)buf * 128 * KS_STRIDE + key * KS_STRIDE + d8 * 8);
    }
  };

  // --- Q strip (pre-scaled by 1/sqrt(D)): 16 x 64 = 128 chunks ---
  if (tid < 128) {
    int r = tid >> 3, d8 = tid & 7;
    async_cp16(Q16 + ((size_t)bh * L_ + q0 + r) * D_ + d8 * 8,
               Qs + r * QS_STRIDE + d8 * 8);
  }
  stageK(0, 0);

  // --- phase 1: S = Q @ K^T, masked, into LDS strip ---
  int buf = 0;
  for (int kt = 0; kt < L_; kt += 128) {
    async_wait0();
    __syncthreads();
    if (kt + 128 < L_) stageK(buf ^ 1, kt + 128);

    const h16* kb = KV + (size_t)buf * 128 * KS_STRIDE;
    v16h a0 = lds_a_frag(Qs, QS_STRIDE, lane);
    v16h a1 = lds_a_frag(Qs + 32, QS_STRIDE, lane);
    v16h b0 = lds_b_frag(kb + (w * 16) * KS_STRIDE,      KS_STRIDE, lane);
    v16h b1 = lds_b_frag(kb + (w * 16) * KS_STRIDE + 32, KS_STRIDE, lane);
    v8f c = vzero8();
    c = wmma_f16(a0, b0, c);
    c = wmma_f16(a1, b1, c);

    const int n = kt + w * 16 + (lane & 15);
#pragma unroll
    for (int r = 0; r < 8; ++r) {
      const int m = r + ((lane >> 4) << 3);
      const int mk = mask[((size_t)b * L_ + q0 + m) * L_ + n];
      Ss[m * L_ + n] = (mk == 0) ? -1.0e9f : c[r];
    }
    buf ^= 1;
  }
  __syncthreads();

  // --- phase 2: softmax (wave w owns rows 2w, 2w+1) ---
#pragma unroll
  for (int rr = 0; rr < 2; ++rr) {
    const int row = w * 2 + rr;
    float mx = -3.0e38f;
    for (int i = lane; i < L_; i += 32) mx = fmaxf(mx, Ss[row * L_ + i]);
#pragma unroll
    for (int off = 16; off >= 1; off >>= 1) mx = fmaxf(mx, __shfl_xor(mx, off, 32));
    float sum = 0.f;
    for (int i = lane; i < L_; i += 32) sum += __expf(Ss[row * L_ + i] - mx);
#pragma unroll
    for (int off = 16; off >= 1; off >>= 1) sum += __shfl_xor(sum, off, 32);
    if (lane == 0) { rowmax[row] = mx; rowinv[row] = 1.0f / sum; }
  }
  __syncthreads();

  for (int i = tid; i < 16 * L_; i += 256) {
    const int m = i >> 11, n = i & (L_ - 1);
    const float p = __expf(Ss[i] - rowmax[m]) * rowinv[m];
    attOut[((size_t)bh * L_ + q0 + m) * L_ + n] = p;
    Ps[m * PS_STRIDE + n] = (h16)p;
  }
  __syncthreads();   // Ss strip dead from here; reused for V tiles

  // --- phase 3: O = P @ V (waves 0-3 keys 0..1023, waves 4-7 1024..2047) ---
  h16* Vs = (h16*)Ss;  // [2 buf][2 half][64 d][VT_STRIDE]
  auto stageV = [&](int vb, int kt) {  // 2 halves x 64 d x 16 chunks
#pragma unroll
    for (int j = 0; j < 8; ++j) {
      int c = j * 256 + tid;
      int half = c >> 10, rem = c & 1023;
      int d = rem >> 4, kc = rem & 15;
      async_cp16(V16t + ((size_t)bh * D_ + d) * L_ + half * 1024 + kt + kc * 8,
                 Vs + (size_t)(((vb * 2 + half) * 64 + d)) * VT_STRIDE + kc * 8);
    }
  };

  const int d0    = (w & 3) * 16;
  const int kbase = (w >> 2) * 1024;
  v8f o = vzero8();
  stageV(0, 0);
  buf = 0;
  for (int kt = 0; kt < 1024; kt += 128) {
    async_wait0();
    __syncthreads();
    if (kt + 128 < 1024) stageV(buf ^ 1, kt + 128);

    const h16* vb = Vs + (size_t)(((buf * 2 + (w >> 2)) * 64 + d0)) * VT_STRIDE;
#pragma unroll
    for (int kk = 0; kk < 128; kk += 32) {
      v16h a  = lds_a_frag(Ps + kbase + kt + kk, PS_STRIDE, lane);
      v16h bb = lds_b_frag(vb + kk, VT_STRIDE, lane);
      o = wmma_f16(a, bb, o);
    }
    buf ^= 1;
  }

  // reduce the two key-halves via LDS (reuse K region), store f16 values
  float* red = (float*)KV;
  if (w >= 4) {
#pragma unroll
    for (int r = 0; r < 8; ++r) red[((w - 4) * 8 + r) * 32 + lane] = o[r];
  }
  __syncthreads();
  if (w < 4) {
#pragma unroll
    for (int r = 0; r < 8; ++r) {
      o[r] += red[(w * 8 + r) * 32 + lane];
      const int m = r + ((lane >> 4) << 3);
      const int e = h * D_ + d0 + (lane & 15);
      VAL16[((size_t)b * L_ + q0 + m) * E_ + e] = (h16)o[r];
    }
  }
}

// ---------------------------------------------------------------------------
// Host launcher
// ---------------------------------------------------------------------------
extern "C" void kernel_launch(void* const* d_in, const int* in_sizes, int n_in,
                              void* d_out, int out_size, void* d_ws,
                              size_t ws_size, hipStream_t stream) {
  const float* in_q = (const float*)d_in[0];
  const float* in_k = (const float*)d_in[1];
  const float* in_v = (const float*)d_in[2];
  const int*   mask = (const int*)d_in[3];
  const float* Wq = (const float*)d_in[4];  const float* bq = (const float*)d_in[5];
  const float* Wk = (const float*)d_in[6];  const float* bk = (const float*)d_in[7];
  const float* Wv = (const float*)d_in[8];  const float* bv = (const float*)d_in[9];
  const float* Wo = (const float*)d_in[10]; const float* bo = (const float*)d_in[11];

  float* o_out   = (float*)d_out;                       // [B,L,E]
  float* att_out = o_out + (size_t)M_ * E_;             // [B,H,L,L]

  char* ws = (char*)d_ws;
  const size_t MB = (size_t)1 << 20;
  h16* Wtq   = (h16*)(ws + 0 * MB);    // [E,E]^T f16, 2 MiB each
  h16* Wtk   = (h16*)(ws + 2 * MB);
  h16* Wtv   = (h16*)(ws + 4 * MB);
  h16* Wto   = (h16*)(ws + 6 * MB);
  h16* Q16   = (h16*)(ws + 8 * MB);    // [B,H,L,D] f16, 16 MiB each
  h16* K16   = (h16*)(ws + 24 * MB);   // [B,H,L,D]
  h16* V16t  = (h16*)(ws + 40 * MB);   // [B,H,D,L]
  h16* VAL16 = (h16*)(ws + 56 * MB);   // [B*L, E]

  dim3 blk(256);
  dim3 gtr(E_ / 32, E_ / 32);          // (32, 32)
  transpose_w_kernel<<<gtr, blk, 0, stream>>>(Wq, Wtq);
  transpose_w_kernel<<<gtr, blk, 0, stream>>>(Wk, Wtk);
  transpose_w_kernel<<<gtr, blk, 0, stream>>>(Wv, Wtv);
  transpose_w_kernel<<<gtr, blk, 0, stream>>>(Wo, Wto);

  dim3 gproj(E_ / 64, M_ / 128);       // (16, 64)
  gemm_kernel<<<gproj, blk, 0, stream>>>(in_q, nullptr, Wtq, bq, Q16, nullptr,
                                         M_, E_, E_, 0.125f, 0);
  gemm_kernel<<<gproj, blk, 0, stream>>>(in_k, nullptr, Wtk, bk, K16, nullptr,
                                         M_, E_, E_, 1.0f, 0);
  gemm_kernel<<<gproj, blk, 0, stream>>>(in_v, nullptr, Wtv, bv, V16t, nullptr,
                                         M_, E_, E_, 1.0f, 2);

  dim3 gattn(L_ / 16, B_ * H_);        // (128, 64)
  attn_kernel<<<gattn, blk, ATTN_SHMEM, stream>>>(Q16, K16, V16t, mask,
                                                  att_out, VAL16);

  gemm_kernel<<<gproj, blk, 0, stream>>>(nullptr, VAL16, Wto, bo, nullptr,
                                         o_out, M_, E_, E_, 1.0f, 1);
}
